// multihead_self_attention_2482491097901
// MI455X (gfx1250) — matrix operand order repaired and checked by reading, not compiled, not run on hardware
//
#include <hip/hip_runtime.h>
#include <hip/hip_bf16.h>

// ---------------------------------------------------------------------------
// MHA w/ RoPE + causal flash attention for gfx1250 (CDNA5, wave32, WMMA).
// All matmuls go through v_wmma_f32_16x16x32_bf16 (f32 accumulate).
// GEMMs use async global->LDS copies (ASYNCcnt) with double buffering.
// ---------------------------------------------------------------------------

#define D_MODEL 1024
#define NUM_HEADS 16
#define HEAD_DIM 64
#define SEQ 2048
#define BATCH 2
#define MTOT (BATCH * SEQ)       // 4096 rows
#define LN_THETA 9.210340371976184f  // ln(10000)

typedef __attribute__((ext_vector_type(16))) __bf16 v16bf;
typedef __attribute__((ext_vector_type(8)))  float  v8f;
typedef __attribute__((ext_vector_type(4)))  unsigned int u32x4;

union Frag {          // one 16x32 (A) or 32x16 (B) bf16 WMMA operand
    v16bf v;
    u32x4 q[2];
};

static __device__ __forceinline__ v8f wmma_bf16(const Frag& a, const Frag& b, v8f c) {
    return __builtin_amdgcn_wmma_f32_16x16x32_bf16(
        /*neg_a=*/false, a.v, /*neg_b=*/false, b.v,
        /*c_mod=*/(short)0, c, /*reuse_a=*/false, /*reuse_b=*/false);
}

// async 16-byte global -> LDS copy (tracked by ASYNCcnt).
// The truncated flat address of a __shared__ object is its LDS byte offset.
static __device__ __forceinline__ void async_copy_b128(const __bf16* g, __bf16* l) {
    unsigned           lds = (unsigned)(unsigned long long)(uintptr_t)l;
    unsigned long long ga  = (unsigned long long)(uintptr_t)g;
    asm volatile("global_load_async_to_lds_b128 %0, %1, off"
                 :: "v"(lds), "v"(ga) : "memory");
}

// ---------------------------------------------------------------------------
// f32 -> bf16 cast
// ---------------------------------------------------------------------------
__global__ __launch_bounds__(256) void cvt_f32_bf16(const float* __restrict__ in,
                                                    __bf16* __restrict__ out, int n) {
    int i = blockIdx.x * 256 + threadIdx.x;
    if (i < n) out[i] = (__bf16)in[i];
}

// ---------------------------------------------------------------------------
// GEMM: C[M,N] = A[M,K] * Bt[N,K]^T  (bf16 row-major; Bt is the JAX weight
// layout W[n][k], so B-fragments are contiguous rows -- no transpose needed).
// Block = 256 threads (8 waves). Block tile 64(M) x 128(N); wave tile 16x64.
// A(64x32) and B(128x32) K-chunks staged in LDS via double-buffered
// global_load_async_to_lds_b128 (3 x 16B per lane per chunk).
// ---------------------------------------------------------------------------
template <bool OUT_BF16>
__global__ __launch_bounds__(256) void gemm_wmma(const __bf16* __restrict__ A,
                                                 const __bf16* __restrict__ Bt,
                                                 void* __restrict__ Cout,
                                                 int Mdim, int Ndim, int Kdim) {
    __shared__ __attribute__((aligned(16))) __bf16 Asb[2][64 * 32];    //  8 KB
    __shared__ __attribute__((aligned(16))) __bf16 Bsb[2][128 * 32];   // 16 KB

    const int tid  = threadIdx.x;
    const int wave = tid >> 5;
    const int lane = tid & 31;
    const int msub  = wave >> 1;          // 0..3 : which 16-row slice
    const int nhalf = wave & 1;           // 0..1 : which 64-col strip
    const int mbase = blockIdx.x * 64;
    const int nbase = blockIdx.y * 128;

    // per-lane async copy assignment: 1 A row-quarter + 2 B row-quarters
    const int crow = tid >> 2;            // 0..63
    const int ccol = (tid & 3) * 8;       // 0,8,16,24 (bf16 elems; 16B units)
    const __bf16* gA  = A  + (size_t)(mbase + crow) * Kdim + ccol;
    const __bf16* gB0 = Bt + (size_t)(nbase + crow) * Kdim + ccol;
    const __bf16* gB1 = Bt + (size_t)(nbase + 64 + crow) * Kdim + ccol;

    auto issue = [&](int kb, int buf) {
        async_copy_b128(gA  + kb, &Asb[buf][crow * 32 + ccol]);
        async_copy_b128(gB0 + kb, &Bsb[buf][crow * 32 + ccol]);
        async_copy_b128(gB1 + kb, &Bsb[buf][(64 + crow) * 32 + ccol]);
    };

    v8f acc[4] = {};
    const int aks = (lane < 16) ? 0 : 16;   // A K-split per lane-half
    const int bks = (lane < 16) ? 0 : 16;   // B K-split per lane-half
    const int arow = msub * 16 + (lane & 15);

    const int nch = Kdim / 32;
    issue(0, 0);
    for (int c = 0; c < nch; ++c) {
        const int cur = c & 1;
        if (c + 1 < nch) {
            issue((c + 1) * 32, cur ^ 1);
            asm volatile("s_wait_asynccnt 0x3" ::: "memory");  // chunk c resident
        } else {
            asm volatile("s_wait_asynccnt 0x0" ::: "memory");
        }
        __syncthreads();

        const __bf16* Asc = Asb[cur];
        const __bf16* Bsc = Bsb[cur];
        Frag af;
        af.q[0] = *(const u32x4*)(&Asc[arow * 32 + aks]);
        af.q[1] = *(const u32x4*)(&Asc[arow * 32 + 8 + aks]);
#pragma unroll
        for (int nt = 0; nt < 4; ++nt) {
            const int nrow = nhalf * 64 + nt * 16 + (lane & 15);
            const u32x4* bp = (const u32x4*)(&Bsc[nrow * 32 + bks]);
            Frag bf;
            bf.q[0] = bp[0];
            bf.q[1] = bp[1];
            acc[nt] = wmma_bf16(af, bf, acc[nt]);
        }
        __syncthreads();   // safe to overwrite buffer `cur` at iteration c+1
    }

    // epilogue: lane holds col = lane%16, rows r + 8*(lane>=16)
    const int hi = (lane >= 16) ? 8 : 0;
#pragma unroll
    for (int nt = 0; nt < 4; ++nt) {
        const int col = nbase + nhalf * 64 + nt * 16 + (lane & 15);
#pragma unroll
        for (int r = 0; r < 8; ++r) {
            const size_t idx = (size_t)(mbase + msub * 16 + r + hi) * Ndim + col;
            float v = acc[nt][r];
            if (OUT_BF16) ((__bf16*)Cout)[idx] = (__bf16)v;
            else          ((float*)Cout)[idx]  = v;
        }
    }
}

// ---------------------------------------------------------------------------
// RoPE + relayout. Reads Qb/Kb/Vb [b*S+s][h*64+d] bf16, writes:
//   Qr,Kr [b][h][s][64]  (Q pre-scaled by 1/sqrt(64) = 0.125)
//   VT    [b][h][64][s]  (transposed so P*V B-fragments are contiguous)
// ---------------------------------------------------------------------------
__global__ __launch_bounds__(256) void rope_relayout(const __bf16* __restrict__ Qb,
                                                     const __bf16* __restrict__ Kb,
                                                     const __bf16* __restrict__ Vb,
                                                     const int* __restrict__ pos,
                                                     __bf16* __restrict__ Qr,
                                                     __bf16* __restrict__ Kr,
                                                     __bf16* __restrict__ VT) {
    const int idx = blockIdx.x * 256 + threadIdx.x;   // B*H*S*32 threads
    const int j = idx & 31;
    const int s = (idx >> 5) & (SEQ - 1);
    const int h = (idx >> 16) & (NUM_HEADS - 1);
    const int b = idx >> 20;

    const float p    = (float)pos[s];
    const float freq = __expf(-((float)(2 * j) / (float)HEAD_DIM) * LN_THETA);
    float sn, cs;
    __sincosf(p * freq, &sn, &cs);

    const size_t src = (size_t)(b * SEQ + s) * D_MODEL + h * HEAD_DIM + 2 * j;
    const size_t bh  = (size_t)(b * NUM_HEADS + h);
    const size_t dstQ = (bh * SEQ + s) * HEAD_DIM + 2 * j;

    float qe = (float)Qb[src], qo = (float)Qb[src + 1];
    Qr[dstQ]     = (__bf16)((qe * cs - qo * sn) * 0.125f);
    Qr[dstQ + 1] = (__bf16)((qe * sn + qo * cs) * 0.125f);

    float ke = (float)Kb[src], ko = (float)Kb[src + 1];
    Kr[dstQ]     = (__bf16)(ke * cs - ko * sn);
    Kr[dstQ + 1] = (__bf16)(ke * sn + ko * cs);

    float ve = (float)Vb[src], vo = (float)Vb[src + 1];
    VT[(bh * HEAD_DIM + 2 * j) * SEQ + s]     = (__bf16)ve;
    VT[(bh * HEAD_DIM + 2 * j + 1) * SEQ + s] = (__bf16)vo;
}

// ---------------------------------------------------------------------------
// Causal flash attention. One wave owns 16 query rows of one (b,h).
// Scores via 2x WMMA (K-dim 64), online softmax with half-wave shfl_xor
// reductions, P re-fragmented through wave-private LDS, P*V via 4x WMMA.
// Output: Ab bf16 [b*S+s][h*64+d]  (row-major M x D_MODEL for the O-proj).
// ---------------------------------------------------------------------------
__global__ __launch_bounds__(128) void flash_attn(const __bf16* __restrict__ Qr,
                                                  const __bf16* __restrict__ Kr,
                                                  const __bf16* __restrict__ VT,
                                                  __bf16* __restrict__ Ab) {
    __shared__ __attribute__((aligned(16))) __bf16 Plds[4 * 16 * 32];   // 4 KB

    const int tid  = threadIdx.x;
    const int wave = tid >> 5;
    const int lane = tid & 31;
    __bf16* myP = &Plds[wave * 16 * 32];

    const int waveId = blockIdx.x * 4 + wave;          // 0..4095
    const int qt = waveId & 127;                       // S/16 tiles
    const int h  = (waveId >> 7) & (NUM_HEADS - 1);
    const int b  = waveId >> 11;
    const int qbase = qt * 16;

    const size_t bh = (size_t)(b * NUM_HEADS + h);
    const __bf16* Qh = Qr + bh * SEQ * HEAD_DIM;
    const __bf16* Kh = Kr + bh * SEQ * HEAD_DIM;
    const __bf16* Vh = VT + bh * HEAD_DIM * SEQ;

    const int hi  = (lane >= 16) ? 8 : 0;
    const int aks = (lane < 16) ? 0 : 16;
    const int bks = (lane < 16) ? 0 : 16;

    // Q fragments for d-chunks [0,32) and [32,64), resident for the whole loop
    Frag aq[2];
    {
        const int qrow = qbase + (lane & 15);
#pragma unroll
        for (int c = 0; c < 2; ++c) {
            const __bf16* p = Qh + (size_t)qrow * HEAD_DIM + c * 32;
            aq[c].q[0] = *(const u32x4*)(p + aks);
            aq[c].q[1] = *(const u32x4*)(p + 8 + aks);
        }
    }

    float mrow[8], lsum[8];
#pragma unroll
    for (int r = 0; r < 8; ++r) { mrow[r] = -1e30f; lsum[r] = 0.0f; }
    v8f oacc[4] = {};

    for (int jb = 0; jb < qbase + 16; jb += 32) {
        // ---- scores: two 16x16 tiles covering j in [jb, jb+32) ----
        v8f sc[2];
#pragma unroll
        for (int t = 0; t < 2; ++t) {
            const int krow = jb + t * 16 + (lane & 15);
            v8f s = {};
#pragma unroll
            for (int c = 0; c < 2; ++c) {
                const __bf16* p = Kh + (size_t)krow * HEAD_DIM + c * 32 + bks;
                Frag bk;
                bk.q[0] = ((const u32x4*)p)[0];
                bk.q[1] = ((const u32x4*)p)[1];
                s = wmma_bf16(aq[c], bk, s);
            }
            sc[t] = s;
        }

        // ---- causal mask + online softmax update ----
        float al[8];
#pragma unroll
        for (int r = 0; r < 8; ++r) {
            const int i  = qbase + r + hi;
            const int j0 = jb + (lane & 15);
            float s0 = (j0      <= i) ? sc[0][r] : -1e30f;
            float s1 = (j0 + 16 <= i) ? sc[1][r] : -1e30f;
            float rm = fmaxf(s0, s1);
            rm = fmaxf(rm, __shfl_xor(rm, 1, 32));
            rm = fmaxf(rm, __shfl_xor(rm, 2, 32));
            rm = fmaxf(rm, __shfl_xor(rm, 4, 32));
            rm = fmaxf(rm, __shfl_xor(rm, 8, 32));
            const float mnew  = fmaxf(mrow[r], rm);
            const float alpha = __expf(mrow[r] - mnew);
            mrow[r] = mnew;
            const float p0 = __expf(s0 - mnew);
            const float p1 = __expf(s1 - mnew);
            float rs = p0 + p1;
            rs += __shfl_xor(rs, 1, 32);
            rs += __shfl_xor(rs, 2, 32);
            rs += __shfl_xor(rs, 4, 32);
            rs += __shfl_xor(rs, 8, 32);
            lsum[r] = lsum[r] * alpha + rs;
            al[r] = alpha;
            // stash P (bf16) for LDS re-fragmentation (wave-private, in-order DS)
            myP[(r + hi) * 32 + (lane & 15)]      = (__bf16)p0;
            myP[(r + hi) * 32 + 16 + (lane & 15)] = (__bf16)p1;
        }

        // rescale running output
#pragma unroll
        for (int nt = 0; nt < 4; ++nt)
#pragma unroll
            for (int r = 0; r < 8; ++r) oacc[nt][r] *= al[r];

        // ---- P * V: re-fragment P from LDS as the A operand ----
        Frag ap;
        {
            const int m = lane & 15;
            ap.q[0] = *(const u32x4*)(&myP[m * 32 + aks]);
            ap.q[1] = *(const u32x4*)(&myP[m * 32 + 8 + aks]);
        }
#pragma unroll
        for (int nt = 0; nt < 4; ++nt) {
            const int vrow = nt * 16 + (lane & 15);
            const __bf16* p = Vh + (size_t)vrow * SEQ + jb + bks;
            Frag bv;
            bv.q[0] = ((const u32x4*)p)[0];
            bv.q[1] = ((const u32x4*)p)[1];
            oacc[nt] = wmma_bf16(ap, bv, oacc[nt]);
        }
    }

    // ---- normalize + write head-interleaved attention output ----
    float inv[8];
#pragma unroll
    for (int r = 0; r < 8; ++r) inv[r] = 1.0f / lsum[r];
#pragma unroll
    for (int nt = 0; nt < 4; ++nt) {
        const int col = h * HEAD_DIM + nt * 16 + (lane & 15);
#pragma unroll
        for (int r = 0; r < 8; ++r) {
            const int i = qbase + r + hi;
            Ab[(size_t)(b * SEQ + i) * D_MODEL + col] = (__bf16)(oacc[nt][r] * inv[r]);
        }
    }
}

// ---------------------------------------------------------------------------
// launch
// ---------------------------------------------------------------------------
extern "C" void kernel_launch(void* const* d_in, const int* in_sizes, int n_in,
                              void* d_out, int out_size, void* d_ws, size_t ws_size,
                              hipStream_t stream) {
    const float* Wq = (const float*)d_in[0];
    const float* Wk = (const float*)d_in[1];
    const float* Wv = (const float*)d_in[2];
    const float* Wo = (const float*)d_in[3];
    const float* X  = (const float*)d_in[4];
    const int*  pos = (const int*)d_in[5];
    float* out = (float*)d_out;

    size_t off = 0;
    auto wsAlloc = [&](size_t bytes) -> void* {
        void* p = (char*)d_ws + off;
        off += (bytes + 255) & ~(size_t)255;
        return p;
    };
    const size_t WN = (size_t)D_MODEL * D_MODEL;       // 1M
    const size_t XN = (size_t)MTOT * D_MODEL;          // 4M
    __bf16* Xb  = (__bf16*)wsAlloc(XN * 2);
    __bf16* Wqb = (__bf16*)wsAlloc(WN * 2);
    __bf16* Wkb = (__bf16*)wsAlloc(WN * 2);
    __bf16* Wvb = (__bf16*)wsAlloc(WN * 2);
    __bf16* Wob = (__bf16*)wsAlloc(WN * 2);
    __bf16* Qb  = (__bf16*)wsAlloc(XN * 2);
    __bf16* Kb  = (__bf16*)wsAlloc(XN * 2);
    __bf16* Vb  = (__bf16*)wsAlloc(XN * 2);
    __bf16* Qr  = (__bf16*)wsAlloc(XN * 2);
    __bf16* Kr  = (__bf16*)wsAlloc(XN * 2);
    __bf16* VT  = (__bf16*)wsAlloc(XN * 2);
    __bf16* Ab  = (__bf16*)wsAlloc(XN * 2);

    // 1) casts
    cvt_f32_bf16<<<(int)(XN / 256), 256, 0, stream>>>(X,  Xb,  (int)XN);
    cvt_f32_bf16<<<(int)(WN / 256), 256, 0, stream>>>(Wq, Wqb, (int)WN);
    cvt_f32_bf16<<<(int)(WN / 256), 256, 0, stream>>>(Wk, Wkb, (int)WN);
    cvt_f32_bf16<<<(int)(WN / 256), 256, 0, stream>>>(Wv, Wvb, (int)WN);
    cvt_f32_bf16<<<(int)(WN / 256), 256, 0, stream>>>(Wo, Wob, (int)WN);

    // 2) Q/K/V projections (bf16 out)
    dim3 ggrid(MTOT / 64, D_MODEL / 128);
    gemm_wmma<true><<<ggrid, 256, 0, stream>>>(Xb, Wqb, Qb, MTOT, D_MODEL, D_MODEL);
    gemm_wmma<true><<<ggrid, 256, 0, stream>>>(Xb, Wkb, Kb, MTOT, D_MODEL, D_MODEL);
    gemm_wmma<true><<<ggrid, 256, 0, stream>>>(Xb, Wvb, Vb, MTOT, D_MODEL, D_MODEL);

    // 3) RoPE + head-major / V-transposed relayout
    rope_relayout<<<(BATCH * NUM_HEADS * SEQ * 32) / 256, 256, 0, stream>>>(
        Qb, Kb, Vb, pos, Qr, Kr, VT);

    // 4) causal flash attention
    flash_attn<<<(BATCH * NUM_HEADS * (SEQ / 16)) / 4, 128, 0, stream>>>(Qr, Kr, VT, Ab);

    // 5) output projection (f32 out -> d_out)
    gemm_wmma<false><<<ggrid, 256, 0, stream>>>(Ab, Wob, out, MTOT, D_MODEL, D_MODEL);
}
